// resblock_dia_77464030151165
// MI455X (gfx1250) — matrix operand order repaired and checked by reading, not compiled, not run on hardware
//
#include <hip/hip_runtime.h>
#include <hip/hip_bf16.h>

// ---------------------------------------------------------------------------
// Sparse 3D conv residual block (MinkowskiEngine-style) for gfx1250 (MI455X).
// gather -> bf16 WMMA 16x16x32 (f32 accum) -> f32 atomic scatter, twice,
// with bias/ReLU/residual epilogues.
//  - weights for offset k staged in LDS once per block (ds_load for B operands)
//  - feature table (8.4 MB bf16) + f32 accumulator live in the 192 MB L2
//  - contiguous-tail sentinel padding => wave breaks out of its tile loop
// ---------------------------------------------------------------------------

typedef __bf16 bf16_t;
typedef __attribute__((ext_vector_type(16))) __bf16 v16bf;
typedef __attribute__((ext_vector_type(8)))  float  v8f;

#define CCH    64     // channels (Cin == Cout)
#define KOFF   27     // 3x3x3 kernel offsets
#define CWAVES 8      // waves per conv block (256 threads, wave32)
#define GRIDY  64     // tile groups per k-offset

__device__ __forceinline__ void atomic_add_f32(float* p, float v) {
    // hardware global_atomic_add_f32 (no return -> STOREcnt path)
    unsafeAtomicAdd(p, v);
}

// ---- pack: feat f32 -> bf16 table with zero pad row; zero the accumulator --
__global__ void pack_feat_kernel(const float* __restrict__ feat,
                                 bf16_t* __restrict__ featb,
                                 float* __restrict__ acc,
                                 int nrows, int N) {
    int i = blockIdx.x * blockDim.x + threadIdx.x;
    if (i >= nrows * CCH) return;
    int n = i / CCH;
    float v = (n < N) ? feat[i] : 0.f;
    featb[i] = (bf16_t)v;
    acc[i]   = 0.f;
}

// ---- pack W[k][cin][cout] f32 -> WMMA B-operand lane layout, bf16 ----------
// dst element for (k, cin, cout):
//   c2 = cin>>5 (cin chunk of 32), g = (cin>>4)&1, e = cin&15
//   c4 = cout>>4 (cout chunk of 16), lane = g*16 + (cout&15)
//   WL[((k*4 + c4)*2 + c2)*32 + lane][e]
__global__ void pack_w_kernel(const float* __restrict__ W,
                              bf16_t* __restrict__ WL) {
    int i = blockIdx.x * blockDim.x + threadIdx.x;
    if (i >= KOFF * CCH * CCH) return;
    int k    = i / (CCH * CCH);
    int rem  = i % (CCH * CCH);
    int cin  = rem / CCH;
    int cout = rem % CCH;
    int c2   = cin >> 5;
    int g    = (cin >> 4) & 1;
    int e    = cin & 15;
    int c4   = cout >> 4;
    int lane = (g << 4) | (cout & 15);
    size_t dst = ((size_t)((k * 4 + c4) * 2 + c2) * 32 + lane) * 16 + e;
    WL[dst] = (bf16_t)W[i];
}

// ---- gather -> WMMA -> atomic scatter ---------------------------------------
// blockIdx.x = kernel offset k; W[k] staged in LDS; each wave grid-strides
// over 16-row tiles of offset k.
__global__ void __launch_bounds__(CWAVES * 32)
sparse_conv_wmma(const bf16_t* __restrict__ featb,
                 const bf16_t* __restrict__ WL,
                 const int* __restrict__ in_idx,
                 const int* __restrict__ out_idx,
                 float* __restrict__ acc,
                 int M, int N) {
    __shared__ uint4 sW[512];                      // 8 KB: W[k] in B-lane layout

    const int k    = blockIdx.x;
    const int wave = threadIdx.x >> 5;
    const int lane = threadIdx.x & 31;
    const int g    = lane >> 4;                    // lane half (K sub-block)
    const int l16  = lane & 15;

    // ---- cooperative LDS fill: 256 threads x 2 uint4 = 8 KB ----
    {
        const uint4* wk = reinterpret_cast<const uint4*>(WL + (size_t)k * (CCH * CCH));
        sW[threadIdx.x]       = wk[threadIdx.x];
        sW[threadIdx.x + 256] = wk[threadIdx.x + 256];
    }
    __syncthreads();

    const int* ii = in_idx  + (size_t)k * M;
    const int* oi = out_idx + (size_t)k * M;

    const int tileStride = GRIDY * CWAVES;
    const int numTiles   = (M + 15) >> 4;

    for (int tile = blockIdx.y * CWAVES + wave; tile < numTiles; tile += tileStride) {
        const int mbase = tile << 4;

        // lane-coalesced index loads (one b32 each), broadcast via lane-permute
        int mm = mbase + l16;
        int r  = (mm < M) ? ii[mm] : N;            // input row (sentinel N = zeros)
        int ov = (mm < M) ? oi[mm] : N;            // output row

        // padding is a contiguous tail: first-entry sentinel => this and all
        // later tiles of this wave are pure padding.
        if (__shfl(r, 0, 32) == N) break;

        // ---- gather A (16x64 bf16 -> two 16x32 WMMA operands) ----
        const uint4* rowp = reinterpret_cast<const uint4*>(featb + (size_t)r * CCH);
        v16bf a0, a1;
        {
            uint4* a0q = reinterpret_cast<uint4*>(&a0);
            uint4* a1q = reinterpret_cast<uint4*>(&a1);
            a0q[0] = rowp[2 * g];        // ch [16g .. 16g+7]
            a0q[1] = rowp[2 * g + 1];    // ch [16g+8 .. +7]
            a1q[0] = rowp[4 + 2 * g];    // ch [32+16g .. +7]
            a1q[1] = rowp[5 + 2 * g];    // ch [40+16g .. +7]
        }

        // out rows for this lane's D elements (row m = j + 8*g -> source lane)
        int orow[8];
#pragma unroll
        for (int j = 0; j < 8; ++j) orow[j] = __shfl(ov, 8 * g + j, 32);

#pragma unroll
        for (int c4 = 0; c4 < 4; ++c4) {
            // B operands from LDS: 32 contiguous bytes per lane per c2 block
            v16bf b0, b1;
            {
                const int base = c4 * 128 + lane * 2;   // uint4 units
                uint4* b0q = reinterpret_cast<uint4*>(&b0);
                uint4* b1q = reinterpret_cast<uint4*>(&b1);
                b0q[0] = sW[base];
                b0q[1] = sW[base + 1];
                b1q[0] = sW[base + 64];
                b1q[1] = sW[base + 65];
            }

            v8f d = {0.f, 0.f, 0.f, 0.f, 0.f, 0.f, 0.f, 0.f};
            d = __builtin_amdgcn_wmma_f32_16x16x32_bf16(false, a0, false, b0,
                                                        (short)0, d, false, false);
            d = __builtin_amdgcn_wmma_f32_16x16x32_bf16(false, a1, false, b1,
                                                        (short)0, d, false, false);

            const int cout = c4 * 16 + l16;
#pragma unroll
            for (int j = 0; j < 8; ++j) {
                int orj = orow[j];
                if (orj != N)   // skip sentinel scatter (avoids hot dummy row)
                    atomic_add_f32(acc + (size_t)orj * CCH + cout, d[j]);
            }
        }
    }
}

// ---- hidden = bf16(relu(acc + b)); reset acc to 0 for the next conv --------
__global__ void bias_relu_pack_kernel(float* __restrict__ acc,
                                      const float* __restrict__ bias,
                                      bf16_t* __restrict__ hb,
                                      int nrows, int N) {
    int i = blockIdx.x * blockDim.x + threadIdx.x;
    if (i >= nrows * CCH) return;
    int n = i / CCH, c = i % CCH;
    float v = 0.f;
    if (n < N) {
        v = acc[i] + bias[c];
        v = v > 0.f ? v : 0.f;
    }
    hb[i]  = (bf16_t)v;
    acc[i] = 0.f;
}

// ---- out = relu(acc + b2) + feat -------------------------------------------
__global__ void final_kernel(const float* __restrict__ acc,
                             const float* __restrict__ bias,
                             const float* __restrict__ feat,
                             float* __restrict__ out, int N) {
    int i = blockIdx.x * blockDim.x + threadIdx.x;
    if (i >= N * CCH) return;
    int c = i % CCH;
    float v = acc[i] + bias[c];
    v = v > 0.f ? v : 0.f;
    out[i] = v + feat[i];
}

static inline size_t align_up(size_t x, size_t a) { return (x + a - 1) & ~(a - 1); }

extern "C" void kernel_launch(void* const* d_in, const int* in_sizes, int n_in,
                              void* d_out, int out_size, void* d_ws, size_t ws_size,
                              hipStream_t stream) {
    const float* feat    = (const float*)d_in[0];
    const float* W1      = (const float*)d_in[1];
    const float* b1      = (const float*)d_in[2];
    const float* W2      = (const float*)d_in[3];
    const float* b2      = (const float*)d_in[4];
    const int*   in_idx  = (const int*)d_in[5];
    const int*   out_idx = (const int*)d_in[6];

    const int N     = in_sizes[0] / CCH;
    const int M     = in_sizes[5] / KOFF;
    const int nrows = N + 1;                  // +1 sentinel/dummy row

    // workspace carve-up
    char*   p       = (char*)d_ws;
    bf16_t* featb   = (bf16_t*)p; p += align_up((size_t)nrows * CCH * sizeof(bf16_t), 256);
    bf16_t* hiddenb = (bf16_t*)p; p += align_up((size_t)nrows * CCH * sizeof(bf16_t), 256);
    float*  acc     = (float*) p; p += align_up((size_t)nrows * CCH * sizeof(float), 256);
    bf16_t* WL1     = (bf16_t*)p; p += align_up((size_t)KOFF * CCH * CCH * sizeof(bf16_t), 256);
    bf16_t* WL2     = (bf16_t*)p; p += align_up((size_t)KOFF * CCH * CCH * sizeof(bf16_t), 256);
    (void)ws_size;

    const int elems  = nrows * CCH;
    const int welems = KOFF * CCH * CCH;

    // 1) pack features + zero accumulator; pack both weight tensors
    pack_feat_kernel<<<(elems + 255) / 256, 256, 0, stream>>>(feat, featb, acc, nrows, N);
    pack_w_kernel<<<(welems + 255) / 256, 256, 0, stream>>>(W1, WL1);
    pack_w_kernel<<<(welems + 255) / 256, 256, 0, stream>>>(W2, WL2);

    dim3 grid(KOFF, GRIDY);
    dim3 block(CWAVES * 32);

    // 2) conv1: gather-WMMA-scatter into acc
    sparse_conv_wmma<<<grid, block, 0, stream>>>(featb, WL1, in_idx, out_idx, acc, M, N);

    // 3) hidden = relu(acc + b1) as bf16; reset acc
    bias_relu_pack_kernel<<<(elems + 255) / 256, 256, 0, stream>>>(acc, b1, hiddenb, nrows, N);

    // 4) conv2
    sparse_conv_wmma<<<grid, block, 0, stream>>>(hiddenb, WL2, in_idx, out_idx, acc, M, N);

    // 5) out = relu(acc + b2) + feat
    final_kernel<<<(N * CCH + 255) / 256, 256, 0, stream>>>(acc, b2, feat, (float*)d_out, N);
}
